// QuantizedLinear_29867202576384
// MI455X (gfx1250) — matrix operand order repaired and checked by reading, not compiled, not run on hardware
//
#include <hip/hip_runtime.h>

typedef __attribute__((ext_vector_type(16))) _Float16 v16h;
typedef __attribute__((ext_vector_type(8)))  _Float16 v8h;
typedef __attribute__((ext_vector_type(4)))  _Float16 v4h;
typedef __attribute__((ext_vector_type(8)))  float    v8f;
typedef __attribute__((ext_vector_type(4)))  float    v4f;
typedef __attribute__((ext_vector_type(4)))  int      v4i;

namespace {
constexpr int M_DIM  = 8192;    // B * S = 4 * 2048
constexpr int N_DIM  = 11008;   // O
constexpr int K_DIM  = 4096;    // I
constexpr int GROUPS = 32;      // I / 128
constexpr int BM = 128, BN = 128, BK = 64;
constexpr int LDA = BK + 8;     // 72 halves = 144B row stride: 16B-aligned b128 frags
constexpr int LDB = BK + 8;
constexpr int KTILES = K_DIM / BK;  // 64
}

__global__ __launch_bounds__(256)
void qlinear_wmma_f16(const float* __restrict__ X,
                      const int*   __restrict__ WS,
                      const float* __restrict__ SC,
                      const float* __restrict__ QB,
                      float*       __restrict__ OUT)
{
    __shared__ _Float16 sA[BM * LDA];   // A tile, [m][k] row-major, f16
    __shared__ _Float16 sB[BN * LDB];   // dequantized W tile, [n][k] row-major, f16

    const int tid  = threadIdx.x;
    const int wave = tid >> 5;
    const int lane = tid & 31;
    const int hlf  = lane >> 4;     // which 16-lane half of the wave
    const int l16  = lane & 15;
    const int wm   = wave & 3;      // 4 waves along M  -> 32 rows each
    const int wn   = wave >> 2;     // 2 waves along N  -> 64 cols each

    const int m0 = blockIdx.y * BM;
    const int n0 = blockIdx.x * BN;

    v8f acc[2][4] = {};             // eight 16x16 f32 accumulators per wave

    // ---- loop-invariant per-lane 32-bit offsets (uniform base advances with kt),
    //      so the backend can use saddr-form global loads: s64 base + v32 offset ----
    int aOff[8];                    // element offsets into X (fits 32-bit: 33.5M max)
    int bOff[4];                    // element offsets into WS (22.5M max)
    int sOff[4];                    // row offsets into scale / q_bias
    #pragma unroll
    for (int j = 0; j < 8; ++j) {
        const int idx = tid + j * 256;
        aOff[j] = (m0 + (idx >> 4)) * K_DIM + (idx & 15) * 4;
    }
    #pragma unroll
    for (int j = 0; j < 4; ++j) {
        const int idx = tid + j * 256;
        bOff[j] = (n0 + (idx >> 3)) * (K_DIM / 2) + (idx & 7) * 4;
        sOff[j] = (n0 + (idx >> 3)) * GROUPS;
    }

    // ---- staging registers for global->LDS pipeline ----
    v4f   aS[8];
    v4i   bS[4];
    float sS[4], qS[4];

    auto load_tile = [&](int kt) {
        const float* xk = X  + kt * BK;        // uniform, advances per tile
        const int*   wk = WS + kt * (BK / 2);  // uniform, advances per tile
        const int    g  = (kt * BK) >> 7;      // quant group, constant across tile
        #pragma unroll
        for (int j = 0; j < 8; ++j)            // 128 rows x 64 k of fp32 x
            aS[j] = *(const v4f*)(xk + aOff[j]);
        #pragma unroll
        for (int j = 0; j < 4; ++j) {          // 128 rows x 32 packed ints + scale/bias
            bS[j] = *(const v4i*)(wk + bOff[j]);
            sS[j] = SC[sOff[j] + g];
            qS[j] = QB[sOff[j] + g];
        }
    };

    auto stage_to_lds = [&]() {
        #pragma unroll
        for (int j = 0; j < 8; ++j) {
            const int idx = tid + j * 256;
            const int r   = idx >> 4;
            const int c4  = idx & 15;
            v4h h;
            #pragma unroll
            for (int t = 0; t < 4; ++t) h[t] = (_Float16)aS[j][t];
            *(v4h*)(&sA[r * LDA + c4 * 4]) = h;            // 8B-aligned ds_store_b64
        }
        #pragma unroll
        for (int j = 0; j < 4; ++j) {
            const int idx = tid + j * 256;
            const int n   = idx >> 3;
            const int c4  = idx & 7;
            const float s = sS[j], q = qS[j];
            v8h h;
            #pragma unroll
            for (int t = 0; t < 4; ++t) {                  // dequant: nibble*scale+bias
                const int v = bS[j][t];
                h[2 * t]     = (_Float16)((float)(v & 15)        * s + q);  // feature 2i
                h[2 * t + 1] = (_Float16)((float)((v >> 4) & 15) * s + q);  // feature 2i+1
            }
            *(v8h*)(&sB[n * LDB + c4 * 8]) = h;            // 16B-aligned ds_store_b128
        }
    };

    load_tile(0);

    for (int kt = 0; kt < KTILES; ++kt) {
        stage_to_lds();
        __syncthreads();
        if (kt + 1 < KTILES) load_tile(kt + 1);            // HBM latency hidden under WMMAs

        #pragma unroll
        for (int kk = 0; kk < BK; kk += 32) {
            v16h af[2], bf[4];
            // A frag per ISA 16-bit A 16x32 layout: two contiguous 8-half chunks
            #pragma unroll
            for (int mi = 0; mi < 2; ++mi) {
                const int m = wm * 32 + mi * 16 + l16;
                v8h lo = *(const v8h*)(&sA[m * LDA + kk + hlf * 8]);
                v8h hi = *(const v8h*)(&sA[m * LDA + kk + 16 + hlf * 8]);
                af[mi] = __builtin_shufflevector(lo, hi, 0,1,2,3,4,5,6,7,8,9,10,11,12,13,14,15);
            }
            // B frag: same two 8-half K chunks as A for its column (stored as W row)
            #pragma unroll
            for (int ni = 0; ni < 4; ++ni) {
                const int n = wn * 64 + ni * 16 + l16;
                v8h b0 = *(const v8h*)(&sB[n * LDB + kk + hlf * 8]);
                v8h b1 = *(const v8h*)(&sB[n * LDB + kk + 16 + hlf * 8]);
                bf[ni] = __builtin_shufflevector(b0, b1, 0,1,2,3,4,5,6,7,8,9,10,11,12,13,14,15);
            }
            #pragma unroll
            for (int mi = 0; mi < 2; ++mi)
                #pragma unroll
                for (int ni = 0; ni < 4; ++ni)
                    acc[mi][ni] = __builtin_amdgcn_wmma_f32_16x16x32_f16(
                        false, af[mi], false, bf[ni],
                        (short)0, acc[mi][ni], false, false);
        }
        __syncthreads();
    }

    // C/D layout: VGPR r, lanes 0-15 -> M=r, lanes 16-31 -> M=8+r, N = lane&15
    #pragma unroll
    for (int mi = 0; mi < 2; ++mi) {
        #pragma unroll
        for (int ni = 0; ni < 4; ++ni) {
            const int n     = n0 + wn * 64 + ni * 16 + l16;
            const int mBase = m0 + wm * 32 + mi * 16 + hlf * 8;
            #pragma unroll
            for (int r = 0; r < 8; ++r)
                OUT[(size_t)(mBase + r) * N_DIM + n] = acc[mi][ni][r];
        }
    }
}

extern "C" void kernel_launch(void* const* d_in, const int* in_sizes, int n_in,
                              void* d_out, int out_size, void* d_ws, size_t ws_size,
                              hipStream_t stream) {
    const float* x  = (const float*)d_in[0];   // [4, 2048, 4096] f32
    const int*   ws = (const int*)d_in[1];     // [11008, 2048] packed nibbles
    const float* sc = (const float*)d_in[2];   // [11008, 32]
    const float* qb = (const float*)d_in[3];   // [11008, 32]
    float* out = (float*)d_out;                // [4, 2048, 11008] f32

    dim3 grid(N_DIM / BN, M_DIM / BM);         // (86, 64)
    qlinear_wmma_f16<<<grid, dim3(256), 0, stream>>>(x, ws, sc, qb, out);

    (void)in_sizes; (void)n_in; (void)out_size; (void)d_ws; (void)ws_size;
}
